// ExLayers_61546881352017
// MI455X (gfx1250) — matrix operand order repaired and checked by reading, not compiled, not run on hardware
//
#include <hip/hip_runtime.h>
#include <hip/hip_bf16.h>
#include <math.h>

typedef __attribute__((ext_vector_type(2)))  float    v2f;
typedef __attribute__((ext_vector_type(8)))  float    v8f;
typedef __attribute__((ext_vector_type(8)))  _Float16 v8h;
typedef __attribute__((ext_vector_type(16))) _Float16 v16h;

#define BB    2048   // batch
#define LL    256    // sequence length
#define DM    2      // d_model
#define DFF   64     // feed-forward dim
#define NL    3      // layers
#define GATEW 13     // gate window
#define PP    4      // polynomial terms
#define NPTS  10     // polynomial control points
#define NT    512    // threads per block = 16 wave32

__device__ __forceinline__ float cubic_w(float d) {
  // PyTorch bicubic kernel, a = -0.75
  const float a = -0.75f;
  float ad = fabsf(d);
  float w1 = ((a + 2.f) * ad - (a + 3.f)) * ad * ad + 1.f;
  float w2 = (((ad - 5.f) * ad + 8.f) * ad - 4.f) * a;
  return ad <= 1.f ? w1 : (ad < 2.f ? w2 : 0.f);
}

__device__ __forceinline__ int clampi(int v, int lo, int hi) {
  return v < lo ? lo : (v > hi ? hi : v);
}

__global__ __launch_bounds__(NT, 1) void fused_transformer_gate(
    const float* __restrict__ spec,
    const float* __restrict__ emb_w, const float* __restrict__ emb_b,
    const float* __restrict__ wqkv,  const float* __restrict__ bqkv,
    const float* __restrict__ wo,    const float* __restrict__ bo,
    const float* __restrict__ ln1w,  const float* __restrict__ ln1b,
    const float* __restrict__ w1,    const float* __restrict__ b1,
    const float* __restrict__ w2,    const float* __restrict__ b2,
    const float* __restrict__ ln2w,  const float* __restrict__ ln2b,
    const float* __restrict__ head_w,const float* __restrict__ head_b,
    const float* __restrict__ poly_w,
    float* __restrict__ out)
{
  const int b    = blockIdx.x;
  const int tid  = threadIdx.x;
  const int wid  = tid >> 5;     // wave id (0..15) -> query tile
  const int lane = tid & 31;
  const int h    = lane >> 4;    // half of wave
  const int n16  = lane & 15;
  const int tok  = tid >> 1;     // FFN: 2 threads per token
  const int part = tid & 1;

  // ---- LDS ----
  __shared__ float    sW[1062];                        // all per-layer weights
  __shared__ float    s_spec[LL];
  __shared__ float    s_x[LL][DM];
  __shared__ v2f      s_q2[LL];                        // packed (q0,q1), pre-scaled
  __shared__ v2f      s_k2[LL];                        // packed (k0,k1)
  __shared__ __align__(16) _Float16 s_vT[DM][LL];      // V transposed, f16 planes
  __shared__ float    s_ao[LL][DM];
  __shared__ float    s_sp[LL + GATEW - 1];
  __shared__ __align__(16) _Float16 s_p[16][16][32];   // per-wave prob transpose

  enum { WQKV=0, BQKV=36, WOW=54, BOW=66, L1W=72, L1B=78,
         W1W=84, B1W=468, W2W=660, B2W=1044, L2W=1050, L2B=1056 };
  for (int i = tid; i < 36;  i += NT) sW[WQKV+i] = wqkv[i];
  for (int i = tid; i < 18;  i += NT) sW[BQKV+i] = bqkv[i];
  for (int i = tid; i < 12;  i += NT) sW[WOW+i]  = wo[i];
  for (int i = tid; i < 6;   i += NT) sW[BOW+i]  = bo[i];
  for (int i = tid; i < 6;   i += NT) sW[L1W+i]  = ln1w[i];
  for (int i = tid; i < 6;   i += NT) sW[L1B+i]  = ln1b[i];
  for (int i = tid; i < 384; i += NT) sW[W1W+i]  = w1[i];
  for (int i = tid; i < 192; i += NT) sW[B1W+i]  = b1[i];
  for (int i = tid; i < 384; i += NT) sW[W2W+i]  = w2[i];
  for (int i = tid; i < 6;   i += NT) sW[B2W+i]  = b2[i];
  for (int i = tid; i < 6;   i += NT) sW[L2W+i]  = ln2w[i];
  for (int i = tid; i < 6;   i += NT) sW[L2B+i]  = ln2b[i];

  // ---- embedding + sinusoidal PE (dm=2 -> div=[1] -> pe=[sin l, cos l]) ----
  if (tid < LL) {
    float sv = spec[b * LL + tid];
    s_spec[tid] = sv;
    float p = (float)tid;
    s_x[tid][0] = sv * emb_w[0] + emb_b[0] + __sinf(p);
    s_x[tid][1] = sv * emb_w[1] + emb_b[1] + __cosf(p);
  }
  __syncthreads();

  const float rs2 = 0.70710678118654752f;   // 1/sqrt(dm)

  // common zero fragments
  v2f zz;  zz[0] = 0.f; zz[1] = 0.f;
  v8f cz;
  #pragma unroll
  for (int r = 0; r < 8; ++r) cz[r] = 0.f;
  v8h hz;
  #pragma unroll
  for (int i = 0; i < 8; ++i) hz[i] = (_Float16)0.f;

  for (int li = 0; li < NL; ++li) {
    // ---- QKV projection (2 -> 6), q pre-scaled by 1/sqrt(d) ----
    const float* Wq = sW + WQKV + li * 12;  // [6][2]
    const float* Bq = sW + BQKV + li * 6;
    if (tid < LL) {
      float x0 = s_x[tid][0], x1 = s_x[tid][1];
      v2f qq, kk;
      qq[0] = (x0*Wq[0]  + x1*Wq[1]  + Bq[0]) * rs2;
      qq[1] = (x0*Wq[2]  + x1*Wq[3]  + Bq[1]) * rs2;
      kk[0] =  x0*Wq[4]  + x1*Wq[5]  + Bq[2];
      kk[1] =  x0*Wq[6]  + x1*Wq[7]  + Bq[3];
      s_q2[tid] = qq;
      s_k2[tid] = kk;
      s_vT[0][tid] = (_Float16)(x0*Wq[8]  + x1*Wq[9]  + Bq[4]);
      s_vT[1][tid] = (_Float16)(x0*Wq[10] + x1*Wq[11] + Bq[5]);
    }
    __syncthreads();

    // ---- scores: V_WMMA_F32_16X16X4_F32, dm=2 padded to K=4 (f32 precision).
    // A layout (f32 16x4): lanes 0-15 hold M=lane, K=0/1; lanes 16-31 K=2/3 (pad=0).
    // B layout (f32 4x16): lanes 0-15 col N=lane, K=0/1; lanes 16-31 K=2/3 (pad=0).
    v2f aQ = (h == 0) ? s_q2[wid*16 + n16] : zz;
    v2f kw[16];
    #pragma unroll
    for (int kt = 0; kt < 16; ++kt) kw[kt] = s_k2[kt*16 + n16];

    v8f acc[16];
    #pragma unroll
    for (int kt = 0; kt < 16; ++kt) {
      v2f bK = (h == 0) ? kw[kt] : zz;
      acc[kt] = __builtin_amdgcn_wmma_f32_16x16x4_f32(false, aQ, false, bK,
                                                      (short)0, cz, false, false);
    }

    // ---- softmax (deferred normalization): keep unnormalized exponentials,
    // remember 1/rowsum and apply it to the 16x2 output after P@V.
    float inv[8];
    #pragma unroll
    for (int r = 0; r < 8; ++r) {
      float m = -3.0e38f;
      #pragma unroll
      for (int kt = 0; kt < 16; ++kt) m = fmaxf(m, acc[kt][r]);
      #pragma unroll
      for (int off = 1; off < 16; off <<= 1) m = fmaxf(m, __shfl_xor(m, off, 32));
      float ssum = 0.f;
      #pragma unroll
      for (int kt = 0; kt < 16; ++kt) {
        float e = __expf(acc[kt][r] - m);
        acc[kt][r] = e;
        ssum += e;
      }
      #pragma unroll
      for (int off = 1; off < 16; off <<= 1) ssum += __shfl_xor(ssum, off, 32);
      inv[r] = 1.f / ssum;
    }

    // ---- E @ V: 8 accumulating f16 WMMAs over 32-key chunks.
    // Exps are in D-layout (lane = column); transpose each chunk through LDS.
    v8f oacc = cz;
    #pragma unroll
    for (int c = 0; c < 8; ++c) {
      // scatter this chunk's exponentials (rows r+8h, cols n16 / 16+n16)
      #pragma unroll
      for (int r = 0; r < 8; ++r) {
        s_p[wid][r + 8*h][n16]      = (_Float16)acc[2*c  ][r];
        s_p[wid][r + 8*h][16 + n16] = (_Float16)acc[2*c+1][r];
      }
      // B fragment: V chunk 32x16, column N = feature dim (only 0,1 nonzero);
      // lanes 0-15 hold K=0..15, lanes 16-31 K=16..31 -> two b128 loads.
      const v8h* vp = (const v8h*)&s_vT[(n16 < DM) ? n16 : 0][c*32 + h*16];
      v8h va = vp[0];
      v8h vb = vp[1];
      va = (n16 < DM) ? va : hz;
      vb = (n16 < DM) ? vb : hz;
      v16h bV = __builtin_shufflevector(va, vb,
                 0,1,2,3,4,5,6,7,8,9,10,11,12,13,14,15);
      asm volatile("s_wait_dscnt 0" ::: "memory");   // staging stores visible
      // A fragment: row M=n16; K slots i -> 16h+i (contiguous)
      const v8h* pr = (const v8h*)&s_p[wid][n16][0];
      v8h lo = pr[2*h];
      v8h hi = pr[2*h + 1];
      v16h aP = __builtin_shufflevector(lo, hi,
                 0,1,2,3,4,5,6,7,8,9,10,11,12,13,14,15);
      oacc = __builtin_amdgcn_wmma_f32_16x16x32_f16(false, aP, false, bV,
                                                    (short)0, oacc, false, false);
    }
    if (n16 < DM) {
      #pragma unroll
      for (int r = 0; r < 8; ++r)
        s_ao[wid*16 + r + 8*h][n16] = oacc[r] * inv[r];
    }
    __syncthreads();

    // ---- residual + Wo + LN1, then FFN (2->64->2) + LN2.
    // Two threads per token; each partner does 32 of 64 hidden units,
    // partials combined with an adjacent-lane shuffle.
    const float* Wo  = sW + WOW + li*4;
    const float* Bo  = sW + BOW + li*2;
    const float* G1  = sW + L1W + li*2;   const float* Be1 = sW + L1B + li*2;
    const float* W1p = sW + W1W + li*128; const float* B1p = sW + B1W + li*64;
    const float* W2p = sW + W2W + li*128; const float* B2p = sW + B2W + li*2;
    const float* G2  = sW + L2W + li*2;   const float* Be2 = sW + L2B + li*2;
    {
      float a0 = s_ao[tok][0], a1 = s_ao[tok][1];
      float x0 = s_x[tok][0] + a0*Wo[0] + a1*Wo[1] + Bo[0];
      float x1 = s_x[tok][1] + a0*Wo[2] + a1*Wo[3] + Bo[1];
      float mu = 0.5f*(x0 + x1);
      float d0 = x0 - mu, d1 = x1 - mu;
      float rstd = rsqrtf(0.5f*(d0*d0 + d1*d1) + 1e-5f);
      x0 = d0*rstd*G1[0] + Be1[0];
      x1 = d1*rstd*G1[1] + Be1[1];
      float f0 = 0.f, f1 = 0.f;
      const int j0 = part * (DFF/2);
      #pragma unroll 8
      for (int j = j0; j < j0 + DFF/2; ++j) {
        float hj = fmaxf(x0*W1p[2*j] + x1*W1p[2*j+1] + B1p[j], 0.f);
        f0 += hj * W2p[j];
        f1 += hj * W2p[DFF + j];
      }
      f0 += __shfl_xor(f0, 1, 32);   // combine partner's 32 hidden units
      f1 += __shfl_xor(f1, 1, 32);
      if (part == 0) {
        float y0 = x0 + f0 + B2p[0];
        float y1 = x1 + f1 + B2p[1];
        mu = 0.5f*(y0 + y1);
        d0 = y0 - mu; d1 = y1 - mu;
        rstd = rsqrtf(0.5f*(d0*d0 + d1*d1) + 1e-5f);
        s_x[tok][0] = d0*rstd*G2[0] + Be2[0];
        s_x[tok][1] = d1*rstd*G2[1] + Be2[1];
      }
    }
    __syncthreads();
  }

  // ---- head -> energy position ----
  float ep = 0.f;
  if (tid < LL) {
    float x0 = s_x[tid][0], x1 = s_x[tid][1];
    ep = tanhf(x0*head_w[0] + x1*head_w[1] + head_b[0]);
  }
  // ---- bicubic stretch-pad of spec: 256 -> 268 (align_corners) ----
  if (tid < LL + GATEW - 1) {
    float src = (float)tid * ((float)(LL - 1) / (float)(LL + GATEW - 2));
    float fi0 = floorf(src);
    int   i0  = (int)fi0;
    float tf  = src - fi0;
    float a = 0.f;
    #pragma unroll
    for (int kk = -1; kk <= 2; ++kk) {
      int idx = clampi(i0 + kk, 0, LL - 1);
      a += s_spec[idx] * cubic_w(tf - (float)kk);
    }
    s_sp[tid] = a;
  }
  __syncthreads();

  // ---- two-hot gather + interpolated polynomial correction ----
  if (tid < LL) {
    float e = ep * (GATEW * 0.5f) + (GATEW * 0.5f);
    e = fminf(fmaxf(e, 0.f), (float)(GATEW - 1) - 1e-6f);
    float lo = floorf(e);
    float fr = e - lo;
    int   il = (int)lo;
    float s = s_sp[tid + il] * (1.f - fr) + s_sp[tid + il + 1] * fr;

    // cubic interp of poly_w rows: 10 control points -> position tid
    float src = (float)tid * ((float)(NPTS - 1) / (float)(LL - 1));
    float fi0 = floorf(src);
    int   i0  = (int)fi0;
    float tf  = src - fi0;
    float wk[4]; int ik[4];
    #pragma unroll
    for (int kk = 0; kk < 4; ++kk) {
      ik[kk] = clampi(i0 + kk - 1, 0, NPTS - 1);
      wk[kk] = cubic_w(tf - (float)(kk - 1));
    }
    float accp = 0.f, term = 1.f;
    #pragma unroll
    for (int i = 0; i < PP; ++i) {
      const float* pr = poly_w + i * NPTS;
      float pw = pr[ik[0]]*wk[0] + pr[ik[1]]*wk[1] + pr[ik[2]]*wk[2] + pr[ik[3]]*wk[3];
      accp += term * pw;
      term *= s;
    }
    out[b * LL + tid] = s + accp;
  }
}

extern "C" void kernel_launch(void* const* d_in, const int* in_sizes, int n_in,
                              void* d_out, int out_size, void* d_ws, size_t ws_size,
                              hipStream_t stream) {
  const float* spec   = (const float*)d_in[0];
  const float* emb_w  = (const float*)d_in[1];
  const float* emb_b  = (const float*)d_in[2];
  const float* wqkv   = (const float*)d_in[3];
  const float* bqkv   = (const float*)d_in[4];
  const float* wo     = (const float*)d_in[5];
  const float* bo     = (const float*)d_in[6];
  const float* ln1w   = (const float*)d_in[7];
  const float* ln1b   = (const float*)d_in[8];
  const float* w1     = (const float*)d_in[9];
  const float* b1     = (const float*)d_in[10];
  const float* w2     = (const float*)d_in[11];
  const float* b2     = (const float*)d_in[12];
  const float* ln2w   = (const float*)d_in[13];
  const float* ln2b   = (const float*)d_in[14];
  const float* head_w = (const float*)d_in[15];
  const float* head_b = (const float*)d_in[16];
  const float* poly_w = (const float*)d_in[17];
  float* outp = (float*)d_out;

  fused_transformer_gate<<<BB, NT, 0, stream>>>(
      spec, emb_w, emb_b, wqkv, bqkv, wo, bo, ln1w, ln1b,
      w1, b1, w2, b2, ln2w, ln2b, head_w, head_b, poly_w, outp);
}
